// GPT2Attention_11115375362451
// MI455X (gfx1250) — matrix operand order repaired and checked by reading, not compiled, not run on hardware
//
#include <hip/hip_runtime.h>

typedef __attribute__((ext_vector_type(16))) __bf16        v16bf;
typedef __attribute__((ext_vector_type(8)))  float         v8f;
typedef __attribute__((ext_vector_type(8)))  unsigned int  v8u;
typedef __attribute__((ext_vector_type(4)))  unsigned int  v4u;

#define WMMA_BF16(a, b, c) \
  __builtin_amdgcn_wmma_f32_16x16x32_bf16(false, (a), false, (b), (short)0, (c), false, false)

static constexpr int BB = 2, S = 2048, D = 1024, H = 16, Dh = 64, N3 = 3072;

static __device__ __forceinline__ unsigned short f2bf(float f) {
  unsigned int u = __float_as_uint(f);
  unsigned int r = (u + 0x7FFFu + ((u >> 16) & 1u)) >> 16;
  return (unsigned short)r;
}
static __device__ __forceinline__ unsigned pk2(float lo, float hi) {
  return (unsigned)f2bf(lo) | ((unsigned)f2bf(hi) << 16);
}

// Async copy 16B/lane global -> LDS (raw bytes), tracked by ASYNCcnt.
static __device__ __forceinline__ void async_b128(unsigned lds_off, const void* gptr) {
  asm volatile("global_load_async_to_lds_b128 %0, %1, off"
               :: "v"(lds_off), "v"((unsigned long long)(uintptr_t)gptr)
               : "memory");
}
static __device__ __forceinline__ void wait_async0() {
  asm volatile("s_wait_asynccnt 0" ::: "memory");
}

// A-fragment (16x32 bf16, MxK) from LDS row-major bf16, row stride RS shorts.
// Lane<16: row=lane, K slots {0..15}; lane>=16: row=lane-16, slots {16..31}.
template <int RS>
static __device__ __forceinline__ v16bf load_a_lds(const unsigned short* base, int rowbase) {
  const int lane = threadIdx.x & 31;
  const unsigned short* rp = base + (rowbase + (lane & 15)) * RS + ((lane >> 4) ? 16 : 0);
  v8u t;
#pragma unroll
  for (int j = 0; j < 8; ++j) {
    int k = 2 * j;
    t[j] = *(const unsigned int*)(rp + k);
  }
  return __builtin_bit_cast(v16bf, t);
}

// B-fragment (32x16 bf16, KxN) from LDS packed pairs Bs[k2][n]={B[2k2][n],B[2k2+1][n]}.
template <int RS>
static __device__ __forceinline__ v16bf load_b_lds(const unsigned int* base, int nbase) {
  const int lane = threadIdx.x & 31;
  const unsigned int* p = base + ((lane >> 4) ? 8 : 0) * RS + nbase + (lane & 15);
  v8u t;
#pragma unroll
  for (int j = 0; j < 8; ++j) t[j] = p[j * RS];
  return __builtin_bit_cast(v16bf, t);
}

// 16 contiguous bf16 from LDS (16B-aligned) as two b128 loads.
static __device__ __forceinline__ v16bf lds_v16(const unsigned short* p) {
  v4u lo = *(const v4u*)p;
  v4u hi = *(const v4u*)(p + 8);
  v8u t;
  t[0] = lo[0]; t[1] = lo[1]; t[2] = lo[2]; t[3] = lo[3];
  t[4] = hi[0]; t[5] = hi[1]; t[6] = hi[2]; t[7] = hi[3];
  return __builtin_bit_cast(v16bf, t);
}

// ---------------------------------------------------------------------------
// Kernel 1: qkv = hidden @ w_attn + b_attn ; scatter to Q/K/V bf16 (head-split).
// Q is pre-scaled by 1/sqrt(Dh)=0.125 (exact in bf16) to slim the attention loop.
// Software-pipelined: vectorized global->regs prefetch, double-buffered LDS.
// ---------------------------------------------------------------------------
__global__ __launch_bounds__(256) void qkv_kernel(
    const float* __restrict__ hid, const float* __restrict__ wA,
    const float* __restrict__ bA, unsigned short* __restrict__ qw,
    unsigned short* __restrict__ kw, unsigned short* __restrict__ vw) {
  __shared__ unsigned short As[2][128 * 40];
  __shared__ unsigned int   Bs[2][16 * 132];
  const int tid = threadIdx.x;
  const int wid = tid >> 5, lane = tid & 31, half = lane >> 4, l15 = lane & 15;
  const int n0 = blockIdx.x * 128, m0 = blockIdx.y * 128;
  const int wm = (wid >> 2) * 64, wn = (wid & 3) * 32;

  v8f acc[4][2];
#pragma unroll
  for (int mt = 0; mt < 4; ++mt)
#pragma unroll
    for (int nt = 0; nt < 2; ++nt)
#pragma unroll
      for (int j = 0; j < 8; ++j) acc[mt][nt][j] = 0.0f;

  float4 abuf[4], blo4[2], bhi4[2];
  auto gload = [&](int kk) {
#pragma unroll
    for (int i = 0; i < 4; ++i) {  // A tile: 128 rows x 8 float4
      int e = i * 256 + tid, r = e >> 3, c4 = (e & 7) * 4;
      abuf[i] = *(const float4*)&hid[(m0 + r) * D + kk + c4];
    }
#pragma unroll
    for (int i = 0; i < 2; ++i) {  // B tile: 16 k2-rows x 32 float4 (x2 rows)
      int e = i * 256 + tid, k2 = e >> 5, n4 = (e & 31) * 4;
      blo4[i] = *(const float4*)&wA[(kk + 2 * k2) * N3 + n0 + n4];
      bhi4[i] = *(const float4*)&wA[(kk + 2 * k2 + 1) * N3 + n0 + n4];
    }
  };
  auto lstore = [&](int p) {
#pragma unroll
    for (int i = 0; i < 4; ++i) {
      int e = i * 256 + tid, r = e >> 3, c4 = (e & 7) * 4;
      unsigned long long pk =
          (unsigned long long)pk2(abuf[i].x, abuf[i].y) |
          ((unsigned long long)pk2(abuf[i].z, abuf[i].w) << 32);
      *(unsigned long long*)&As[p][r * 40 + c4] = pk;  // 8B-aligned b64 store
    }
#pragma unroll
    for (int i = 0; i < 2; ++i) {
      int e = i * 256 + tid, k2 = e >> 5, n4 = (e & 31) * 4;
      v4u pk;
      pk[0] = pk2(blo4[i].x, bhi4[i].x);
      pk[1] = pk2(blo4[i].y, bhi4[i].y);
      pk[2] = pk2(blo4[i].z, bhi4[i].z);
      pk[3] = pk2(blo4[i].w, bhi4[i].w);
      *(v4u*)&Bs[p][k2 * 132 + n4] = pk;  // 16B-aligned b128 store
    }
  };

  gload(0);
  for (int kk = 0, it = 0; kk < D; kk += 32, ++it) {
    const int p = it & 1;
    lstore(p);
    __syncthreads();
    if (kk + 32 < D) gload(kk + 32);

    v16bf af[4], bfr[2];
#pragma unroll
    for (int mt = 0; mt < 4; ++mt) af[mt] = load_a_lds<40>(As[p], wm + mt * 16);
#pragma unroll
    for (int nt = 0; nt < 2; ++nt) bfr[nt] = load_b_lds<132>(Bs[p], wn + nt * 16);
#pragma unroll
    for (int mt = 0; mt < 4; ++mt)
#pragma unroll
      for (int nt = 0; nt < 2; ++nt) acc[mt][nt] = WMMA_BF16(af[mt], bfr[nt], acc[mt][nt]);
  }

#pragma unroll
  for (int mt = 0; mt < 4; ++mt) {
#pragma unroll
    for (int nt = 0; nt < 2; ++nt) {
      int col = n0 + wn + nt * 16 + l15;
      float bias = bA[col];
      int which = col >> 10, d0 = col & 1023, h = d0 >> 6, dh = d0 & 63;
#pragma unroll
      for (int j = 0; j < 8; ++j) {
        int row = m0 + wm + mt * 16 + j + 8 * half;
        int b = row >> 11, s = row & 2047;
        float x = acc[mt][nt][j] + bias;
        if (which == 0) {  // Q pre-scaled by 1/sqrt(Dh) (exact: power of two)
          qw[((b * H + h) * S + s) * Dh + dh] = f2bf(x * 0.125f);
        } else if (which == 1) {
          kw[((b * H + h) * S + s) * Dh + dh] = f2bf(x);
        } else {  // V pair-packed over key dim: u32[(bh)][s/2][dh], low half = even s
          vw[((((b * H + h) * (S / 2) + (s >> 1)) * Dh + dh) << 1) + (s & 1)] = f2bf(x);
        }
      }
    }
  }
}

// ---------------------------------------------------------------------------
// Kernel 2: flash attention. K/V tiles staged once per workgroup into LDS via
// async global->LDS copies (ASYNCcnt), double-buffered. One wave = 16 q rows.
// ---------------------------------------------------------------------------
__global__ __launch_bounds__(256) void attn_kernel(
    const unsigned short* __restrict__ qg, const unsigned short* __restrict__ kg,
    const unsigned int* __restrict__ vg, unsigned short* __restrict__ og) {
  __shared__ unsigned short Kt[2][32 * 72];  // 32 keys x 64 bf16, row pad-> 144B
  __shared__ unsigned int   Vt[2][16 * 68];  // 16 pair-rows x 64 u32, pad-> 272B
  __shared__ unsigned short Pl[8][16 * 34];
  const int tid = threadIdx.x;
  const int wid = tid >> 5, lane = tid & 31, half = lane >> 4, l15 = lane & 15;
  const int bh = blockIdx.y, b = bh >> 4, h = bh & 15;
  const int qbase = blockIdx.x * 128 + wid * 16;
  const unsigned short* qp = qg + bh * (S * Dh);
  const unsigned short* kp = kg + bh * (S * Dh);
  const unsigned int*   vp = vg + bh * ((S / 2) * Dh);

  // staging maps: K tile = 32 rows x 8 segs x 16B; V tile = 16 rows x 16 segs x 16B
  const int krow = tid >> 3, kseg = tid & 7;
  const int vrow = tid >> 4, vseg = tid & 15;
  auto issue = [&](int kt, int p) {
    async_b128((unsigned)(uintptr_t)&Kt[p][krow * 72 + kseg * 8],
               kp + (kt + krow) * Dh + kseg * 8);
    async_b128((unsigned)(uintptr_t)&Vt[p][vrow * 68 + vseg * 4],
               vp + ((kt >> 1) + vrow) * Dh + vseg * 4);
  };

  // Q A-fragments for Dh=64 (two 32-wide K chunks), identity slot<->d mapping.
  v16bf qf[2];
#pragma unroll
  for (int f = 0; f < 2; ++f) {
    const unsigned short* rp = qp + (qbase + l15) * Dh + f * 32 + (half ? 16 : 0);
    v8u t;
#pragma unroll
    for (int j = 0; j < 8; ++j) {
      int k = 2 * j;
      t[j] = *(const unsigned int*)(rp + k);
    }
    qf[f] = __builtin_bit_cast(v16bf, t);
  }

  float m[8], l[8];
  v8f acc[4];
#pragma unroll
  for (int j = 0; j < 8; ++j) { m[j] = -1e30f; l[j] = 0.0f; }
#pragma unroll
  for (int dt = 0; dt < 4; ++dt)
#pragma unroll
    for (int j = 0; j < 8; ++j) acc[dt][j] = 0.0f;

  v8f zero;
#pragma unroll
  for (int j = 0; j < 8; ++j) zero[j] = 0.0f;

  issue(0, 0);
  for (int kt = 0, it = 0; kt < S; kt += 32, ++it) {
    const int p = it & 1;
    wait_async0();
    __syncthreads();
    if (kt + 32 < S) issue(kt + 32, p ^ 1);

    // scores = (Q/8) @ K^T, two 16-key N tiles, chained over Dh (K frags from LDS)
    const unsigned short* kr0 = &Kt[p][l15 * 72 + half * 16];
    const unsigned short* kr1 = &Kt[p][(16 + l15) * 72 + half * 16];
    v8f s0 = WMMA_BF16(qf[0], lds_v16(kr0), zero);
    s0 = WMMA_BF16(qf[1], lds_v16(kr0 + 32), s0);
    v8f s1 = WMMA_BF16(qf[0], lds_v16(kr1), zero);
    s1 = WMMA_BF16(qf[1], lds_v16(kr1 + 32), s1);

    float alpha[8];
#pragma unroll
    for (int j = 0; j < 8; ++j) {
      float t = fmaxf(s0[j], s1[j]);
      t = fmaxf(t, __shfl_xor(t, 1, 32));
      t = fmaxf(t, __shfl_xor(t, 2, 32));
      t = fmaxf(t, __shfl_xor(t, 4, 32));
      t = fmaxf(t, __shfl_xor(t, 8, 32));
      float mn = fmaxf(m[j], t);
      alpha[j] = __expf(m[j] - mn);
      m[j] = mn;
      s0[j] = __expf(s0[j] - mn);
      s1[j] = __expf(s1[j] - mn);
      float rs = s0[j] + s1[j];
      rs += __shfl_xor(rs, 1, 32);
      rs += __shfl_xor(rs, 2, 32);
      rs += __shfl_xor(rs, 4, 32);
      rs += __shfl_xor(rs, 8, 32);
      l[j] = l[j] * alpha[j] + rs;
    }

    // P (D-layout) -> per-wave LDS -> A-layout fragment
    unsigned short* pl = &Pl[wid][0];
#pragma unroll
    for (int j = 0; j < 8; ++j) {
      int r = j + 8 * half;
      pl[r * 34 + l15] = f2bf(s0[j]);
      pl[r * 34 + 16 + l15] = f2bf(s1[j]);
    }
    asm volatile("s_wait_dscnt 0" ::: "memory");
    v16bf pf;
    {
      const unsigned short* rp = pl + l15 * 34 + (half ? 16 : 0);
      v8u t;
#pragma unroll
      for (int j = 0; j < 8; ++j) {
        int k = 2 * j;
        t[j] = *(const unsigned int*)(rp + k);
      }
      pf = __builtin_bit_cast(v16bf, t);
    }

    // rescale running output, then acc += P @ V (V fragments from LDS)
#pragma unroll
    for (int dt = 0; dt < 4; ++dt)
#pragma unroll
      for (int j = 0; j < 8; ++j) acc[dt][j] *= alpha[j];
#pragma unroll
    for (int dt = 0; dt < 4; ++dt) {
      const unsigned int* vbp = &Vt[p][(half * 8) * 68 + dt * 16 + l15];
      v8u t;
#pragma unroll
      for (int j = 0; j < 8; ++j) t[j] = vbp[j * 68];
      v16bf vf = __builtin_bit_cast(v16bf, t);
      acc[dt] = WMMA_BF16(pf, vf, acc[dt]);
    }
  }

#pragma unroll
  for (int j = 0; j < 8; ++j) {
    float inv = 1.0f / l[j];
    int srow = qbase + j + 8 * half;
    unsigned short* orow = og + (b * S + srow) * D + h * Dh;
#pragma unroll
    for (int dt = 0; dt < 4; ++dt) orow[dt * 16 + l15] = f2bf(acc[dt][j] * inv);
  }
}

// ---------------------------------------------------------------------------
// Kernel 3: out = attn_out @ w_proj + b_proj (fp32 result), same pipeline.
// ---------------------------------------------------------------------------
__global__ __launch_bounds__(256) void proj_kernel(
    const unsigned short* __restrict__ ao, const float* __restrict__ wP,
    const float* __restrict__ bP, float* __restrict__ out) {
  __shared__ unsigned short As[2][128 * 40];
  __shared__ unsigned int   Bs[2][16 * 132];
  const int tid = threadIdx.x;
  const int wid = tid >> 5, lane = tid & 31, half = lane >> 4, l15 = lane & 15;
  const int n0 = blockIdx.x * 128, m0 = blockIdx.y * 128;
  const int wm = (wid >> 2) * 64, wn = (wid & 3) * 32;

  v8f acc[4][2];
#pragma unroll
  for (int mt = 0; mt < 4; ++mt)
#pragma unroll
    for (int nt = 0; nt < 2; ++nt)
#pragma unroll
      for (int j = 0; j < 8; ++j) acc[mt][nt][j] = 0.0f;

  v4u ar4[2];
  float4 blo4[2], bhi4[2];
  auto gload = [&](int kk) {
#pragma unroll
    for (int i = 0; i < 2; ++i) {  // A tile: 128 rows x 4 x (8 bf16 = 16B)
      int e = i * 256 + tid, r = e >> 2, c8 = (e & 3) * 8;
      ar4[i] = *(const v4u*)&ao[(m0 + r) * D + kk + c8];
    }
#pragma unroll
    for (int i = 0; i < 2; ++i) {
      int e = i * 256 + tid, k2 = e >> 5, n4 = (e & 31) * 4;
      blo4[i] = *(const float4*)&wP[(kk + 2 * k2) * D + n0 + n4];
      bhi4[i] = *(const float4*)&wP[(kk + 2 * k2 + 1) * D + n0 + n4];
    }
  };
  auto lstore = [&](int p) {
#pragma unroll
    for (int i = 0; i < 2; ++i) {
      int e = i * 256 + tid, r = e >> 2, c8 = (e & 3) * 8;
      *(v4u*)&As[p][r * 40 + c8] = ar4[i];  // 16B-aligned (80B row stride)
    }
#pragma unroll
    for (int i = 0; i < 2; ++i) {
      int e = i * 256 + tid, k2 = e >> 5, n4 = (e & 31) * 4;
      v4u pk;
      pk[0] = pk2(blo4[i].x, bhi4[i].x);
      pk[1] = pk2(blo4[i].y, bhi4[i].y);
      pk[2] = pk2(blo4[i].z, bhi4[i].z);
      pk[3] = pk2(blo4[i].w, bhi4[i].w);
      *(v4u*)&Bs[p][k2 * 132 + n4] = pk;
    }
  };

  gload(0);
  for (int kk = 0, it = 0; kk < D; kk += 32, ++it) {
    const int p = it & 1;
    lstore(p);
    __syncthreads();
    if (kk + 32 < D) gload(kk + 32);

    v16bf af[4], bfr[2];
#pragma unroll
    for (int mt = 0; mt < 4; ++mt) af[mt] = load_a_lds<40>(As[p], wm + mt * 16);
#pragma unroll
    for (int nt = 0; nt < 2; ++nt) bfr[nt] = load_b_lds<132>(Bs[p], wn + nt * 16);
#pragma unroll
    for (int mt = 0; mt < 4; ++mt)
#pragma unroll
      for (int nt = 0; nt < 2; ++nt) acc[mt][nt] = WMMA_BF16(af[mt], bfr[nt], acc[mt][nt]);
  }

#pragma unroll
  for (int mt = 0; mt < 4; ++mt) {
#pragma unroll
    for (int nt = 0; nt < 2; ++nt) {
      int col = n0 + wn + nt * 16 + l15;
      float bias = bP[col];
#pragma unroll
      for (int j = 0; j < 8; ++j) {
        int row = m0 + wm + mt * 16 + j + 8 * half;
        out[row * D + col] = acc[mt][nt][j] + bias;
      }
    }
  }
}

// ---------------------------------------------------------------------------
extern "C" void kernel_launch(void* const* d_in, const int* in_sizes, int n_in,
                              void* d_out, int out_size, void* d_ws, size_t ws_size,
                              hipStream_t stream) {
  (void)in_sizes; (void)n_in; (void)out_size; (void)ws_size;
  const float* hid = (const float*)d_in[0];
  const float* wA  = (const float*)d_in[1];
  const float* bA  = (const float*)d_in[2];
  const float* wP  = (const float*)d_in[3];
  const float* bP  = (const float*)d_in[4];
  float* out = (float*)d_out;

  // Workspace layout (bf16 shorts): Q | K | Vpacked | attn_out, 4 x 4,194,304 = 32 MB
  unsigned short* qw = (unsigned short*)d_ws;
  unsigned short* kw = qw + (size_t)BB * H * S * Dh;
  unsigned short* vw = kw + (size_t)BB * H * S * Dh;
  unsigned short* ow = vw + (size_t)BB * H * S * Dh;

  qkv_kernel<<<dim3(N3 / 128, (BB * S) / 128), 256, 0, stream>>>(hid, wA, bA, qw, kw, vw);
  attn_kernel<<<dim3(S / 128, BB * H), 256, 0, stream>>>(qw, kw, (const unsigned int*)vw, ow);
  proj_kernel<<<dim3(D / 128, (BB * S) / 128), 256, 0, stream>>>(ow, wP, bP, out);
}
